// CNN_GNN_Model_78847009620619
// MI455X (gfx1250) — compile-verified, matrix operand order checked by reading, not run on hardware
//
#include <hip/hip_runtime.h>
#include <stdint.h>

// ---------------------------------------------------------------------------
// Problem constants (match reference)
// ---------------------------------------------------------------------------
static const int Bn = 4096;   // batch / graph nodes
static const int Fd = 1536;   // feature dim
static const int Hd = 256;    // hidden dim
static const int Cd = 38;     // classes
static const int Kn = 7;      // knn K
static const int CB = Hd + Fd;   // 1792 concat width
static const int HC = Hd / 2;    // 128 classifier hidden
#define EPSF 1e-5f

// ---------------------------------------------------------------------------
// WMMA vector types (CDNA5 / gfx1250, wave32)
// ---------------------------------------------------------------------------
typedef float  v8f   __attribute__((ext_vector_type(8)));
typedef __bf16 v16bf __attribute__((ext_vector_type(16)));

union U32BF { unsigned u; __bf16 h[2]; };

__device__ inline __bf16 f2bf(float f) { return (__bf16)f; }

// Load a 16x32 bf16 A/B fragment row from LDS (packed-pair layout per ISA
// 7.12.2: lanes 0-15 hold K=0..7,16..23; lanes 16-31 hold K=8..15,24..31).
__device__ inline v16bf ldfrag_bf16(const __bf16* rowp, int kh) {
  const unsigned* p = (const unsigned*)rowp;
  v16bf f;
  U32BF c;
#pragma unroll
  for (int v = 0; v < 4; ++v) {
    c.u = p[v + 4 * kh];
    f[2 * v]     = c.h[0];
    f[2 * v + 1] = c.h[1];
  }
#pragma unroll
  for (int j = 0; j < 4; ++j) {
    c.u = p[8 + j + 4 * kh];
    f[8 + 2 * j] = c.h[0];
    f[9 + 2 * j] = c.h[1];
  }
  return f;
}

// ---------------------------------------------------------------------------
// 1) Feature batch-norm with hi/lo bf16 split:
//    x = bn(features); hi = bf16(x) -> comb[:,256:]; lo = bf16(x - hi) -> xlo.
//    (hi doubles as the Gram "high" operand and the classifier/GEMM1 operand.)
// ---------------------------------------------------------------------------
__global__ __launch_bounds__(256) void bnf_kernel(
    const float* __restrict__ feat, const float* __restrict__ g,
    const float* __restrict__ b, const float* __restrict__ m,
    const float* __restrict__ v, __bf16* __restrict__ comb,
    __bf16* __restrict__ xlo) {
  int idx = blockIdx.x * 256 + threadIdx.x;   // over Bn*Fd
  int c = idx % Fd;
  int r = idx / Fd;
  float xv = (feat[idx] - m[c]) * rsqrtf(v[c] + EPSF) * g[c] + b[c];
  __bf16 hi = f2bf(xv);
  __bf16 lo = f2bf(xv - (float)hi);
  comb[(size_t)r * CB + Hd + c] = hi;
  xlo[idx] = lo;
}

// ---------------------------------------------------------------------------
// 2) Row squared norms of x (recompute BN from features; avoids fp32 x buffer)
// ---------------------------------------------------------------------------
__global__ __launch_bounds__(256) void rowsq_kernel(
    const float* __restrict__ feat, const float* __restrict__ g,
    const float* __restrict__ b, const float* __restrict__ m,
    const float* __restrict__ v, float* __restrict__ sq) {
  __shared__ float red[256];
  int r = blockIdx.x;
  const float* row = feat + (size_t)r * Fd;
  float s = 0.f;
  for (int j = threadIdx.x; j < Fd; j += 256) {
    float t = (row[j] - m[j]) * rsqrtf(v[j] + EPSF) * g[j] + b[j];
    s += t * t;
  }
  red[threadIdx.x] = s;
  __syncthreads();
  for (int st = 128; st > 0; st >>= 1) {
    if (threadIdx.x < st) red[threadIdx.x] += red[threadIdx.x + st];
    __syncthreads();
  }
  if (threadIdx.x == 0) sq[r] = red[0];
}

// ---------------------------------------------------------------------------
// 3) Gram + distances via split-bf16 WMMA:
//    G = hi*hi' + hi*lo' + lo*hi'  (3x V_WMMA_F32_16X16X32_BF16, f32 accum)
//    Block tile 128x64, BK=32, 8 waves (4x2), wave tile 32x32.
//    B operand is x^T so its fragment = rows of x: same LDS layout as A.
// ---------------------------------------------------------------------------
__global__ __launch_bounds__(256) void gram_dist_kernel(
    const __bf16* __restrict__ xhi, int ldh,   // = comb + Hd, lda = CB
    const __bf16* __restrict__ xlo,            // stride Fd
    const float* __restrict__ sq, const float* __restrict__ noise,
    float* __restrict__ dmat) {
  __shared__ __bf16 Ahi[128][40];
  __shared__ __bf16 Alo[128][40];
  __shared__ __bf16 Bhi[64][40];
  __shared__ __bf16 Blo[64][40];

  int tid  = threadIdx.x;
  int lane = tid & 31;
  int wave = tid >> 5;
  int wm = (wave & 3) * 32;
  int wn = (wave >> 2) * 32;
  int bm0 = blockIdx.y * 128;
  int bn0 = blockIdx.x * 64;

  v8f acc[2][2];
#pragma unroll
  for (int i = 0; i < 2; ++i)
#pragma unroll
    for (int j = 0; j < 2; ++j)
#pragma unroll
      for (int e = 0; e < 8; ++e) acc[i][j][e] = 0.f;

  int rl = lane & 15, kh = lane >> 4;

  for (int k0 = 0; k0 < Fd; k0 += 32) {
    // A tiles: 128x32 bf16 each = 2048 u32, 8 u32/thread per matrix
#pragma unroll
    for (int i = 0; i < 8; ++i) {
      int idx = tid + i * 256;
      int r = idx >> 4, cu = idx & 15;
      *(unsigned*)(&Ahi[r][2 * cu]) =
          *(const unsigned*)(xhi + (size_t)(bm0 + r) * ldh + k0 + 2 * cu);
      *(unsigned*)(&Alo[r][2 * cu]) =
          *(const unsigned*)(xlo + (size_t)(bm0 + r) * Fd + k0 + 2 * cu);
    }
    // B tiles: 64x32 bf16 each = 1024 u32, 4 u32/thread per matrix
#pragma unroll
    for (int i = 0; i < 4; ++i) {
      int idx = tid + i * 256;
      int r = idx >> 4, cu = idx & 15;
      *(unsigned*)(&Bhi[r][2 * cu]) =
          *(const unsigned*)(xhi + (size_t)(bn0 + r) * ldh + k0 + 2 * cu);
      *(unsigned*)(&Blo[r][2 * cu]) =
          *(const unsigned*)(xlo + (size_t)(bn0 + r) * Fd + k0 + 2 * cu);
    }
    __syncthreads();

    v16bf a0 = ldfrag_bf16(&Ahi[wm + rl][0], kh);
    v16bf a1 = ldfrag_bf16(&Ahi[wm + 16 + rl][0], kh);
    v16bf b0 = ldfrag_bf16(&Bhi[wn + rl][0], kh);
    v16bf b1 = ldfrag_bf16(&Bhi[wn + 16 + rl][0], kh);
    // hi * hi
    acc[0][0] = __builtin_amdgcn_wmma_f32_16x16x32_bf16(false, a0, false, b0, (short)0, acc[0][0], false, false);
    acc[0][1] = __builtin_amdgcn_wmma_f32_16x16x32_bf16(false, a0, false, b1, (short)0, acc[0][1], false, false);
    acc[1][0] = __builtin_amdgcn_wmma_f32_16x16x32_bf16(false, a1, false, b0, (short)0, acc[1][0], false, false);
    acc[1][1] = __builtin_amdgcn_wmma_f32_16x16x32_bf16(false, a1, false, b1, (short)0, acc[1][1], false, false);
    // hi * lo
    {
      v16bf c0 = ldfrag_bf16(&Blo[wn + rl][0], kh);
      v16bf c1 = ldfrag_bf16(&Blo[wn + 16 + rl][0], kh);
      acc[0][0] = __builtin_amdgcn_wmma_f32_16x16x32_bf16(false, a0, false, c0, (short)0, acc[0][0], false, false);
      acc[0][1] = __builtin_amdgcn_wmma_f32_16x16x32_bf16(false, a0, false, c1, (short)0, acc[0][1], false, false);
      acc[1][0] = __builtin_amdgcn_wmma_f32_16x16x32_bf16(false, a1, false, c0, (short)0, acc[1][0], false, false);
      acc[1][1] = __builtin_amdgcn_wmma_f32_16x16x32_bf16(false, a1, false, c1, (short)0, acc[1][1], false, false);
    }
    // lo * hi
    {
      v16bf d0 = ldfrag_bf16(&Alo[wm + rl][0], kh);
      v16bf d1 = ldfrag_bf16(&Alo[wm + 16 + rl][0], kh);
      acc[0][0] = __builtin_amdgcn_wmma_f32_16x16x32_bf16(false, d0, false, b0, (short)0, acc[0][0], false, false);
      acc[0][1] = __builtin_amdgcn_wmma_f32_16x16x32_bf16(false, d0, false, b1, (short)0, acc[0][1], false, false);
      acc[1][0] = __builtin_amdgcn_wmma_f32_16x16x32_bf16(false, d1, false, b0, (short)0, acc[1][0], false, false);
      acc[1][1] = __builtin_amdgcn_wmma_f32_16x16x32_bf16(false, d1, false, b1, (short)0, acc[1][1], false, false);
    }
    __syncthreads();
  }

  // epilogue: distance transform (C layout: n = lane%16, m = r + 8*(lane/16))
#pragma unroll
  for (int i = 0; i < 2; ++i)
#pragma unroll
    for (int j = 0; j < 2; ++j)
#pragma unroll
      for (int r = 0; r < 8; ++r) {
        int row = bm0 + wm + i * 16 + r + 8 * kh;
        int col = bn0 + wn + j * 16 + rl;
        float g2 = acc[i][j][r];
        float d2 = sq[row] + sq[col] - 2.f * g2;
        d2 = d2 > 0.f ? d2 : 0.f;
        float d = sqrtf(d2) + noise[(size_t)row * Bn + col] * 1e-6f;
        dmat[(size_t)row * Bn + col] = d;
      }
}

// ---------------------------------------------------------------------------
// 4) Single-pass top-(K+1): per-thread register top-8 (unrolled bubble
//    insert), then 8 LDS selection passes over 2048 candidates.
//    Tie-break = lower index first (matches jax.lax.top_k). Drop self.
// ---------------------------------------------------------------------------
__global__ __launch_bounds__(256) void topk_kernel(
    const float* __restrict__ dmat, int* __restrict__ knn) {
  __shared__ float cv[256 * 8];
  __shared__ int   ci[256 * 8];
  __shared__ float sv[256];
  __shared__ int   si[256];
  __shared__ int   result[Kn + 1];
  int r = blockIdx.x, tid = threadIdx.x;
  const float* row = dmat + (size_t)r * Bn;

  float bv[8];
  int   bi[8];
#pragma unroll
  for (int s = 0; s < 8; ++s) { bv[s] = 3.4e38f; bi[s] = 0x7fffffff; }

  for (int j = 0; j < Bn / 256; ++j) {
    int c = tid + j * 256;       // coalesced; columns unique per thread
    float nv = row[c];
    int   ni = c;
#pragma unroll
    for (int s = 0; s < 8; ++s) {
      bool better = (nv < bv[s]) || (nv == bv[s] && ni < bi[s]);
      float tv = better ? bv[s] : nv;
      int   ti = better ? bi[s] : ni;
      bv[s] = better ? nv : bv[s];
      bi[s] = better ? ni : bi[s];
      nv = tv; ni = ti;
    }
  }
#pragma unroll
  for (int s = 0; s < 8; ++s) { cv[tid * 8 + s] = bv[s]; ci[tid * 8 + s] = bi[s]; }
  __syncthreads();

  for (int p = 0; p < Kn + 1; ++p) {
    float mv = 3.4e38f;
    int   mi = 0x7fffffff;
#pragma unroll
    for (int s = 0; s < 8; ++s) {
      float v = cv[tid * 8 + s];
      int   i = ci[tid * 8 + s];
      if (v < mv || (v == mv && i < mi)) { mv = v; mi = i; }
    }
    sv[tid] = mv; si[tid] = mi;
    __syncthreads();
    for (int st = 128; st > 0; st >>= 1) {
      if (tid < st) {
        float v2 = sv[tid + st]; int i2 = si[tid + st];
        if (v2 < sv[tid] || (v2 == sv[tid] && i2 < si[tid])) {
          sv[tid] = v2; si[tid] = i2;
        }
      }
      __syncthreads();
    }
    if (tid == 0) result[p] = si[0];
    __syncthreads();
    int win = result[p];
#pragma unroll
    for (int s = 0; s < 8; ++s)
      if (ci[tid * 8 + s] == win) cv[tid * 8 + s] = 3.4e38f;
    __syncthreads();
  }
  if (tid < Kn) knn[r * Kn + tid] = result[tid + 1];
}

// ---------------------------------------------------------------------------
// 5) Degrees / normalization
// ---------------------------------------------------------------------------
__global__ __launch_bounds__(256) void deg_init_kernel(float* deg) {
  deg[blockIdx.x * 256 + threadIdx.x] = 1.0f;   // self loop
}
__global__ __launch_bounds__(256) void deg_edge_kernel(
    const int* __restrict__ knn, float* __restrict__ deg) {
  int e = blockIdx.x * 256 + threadIdx.x;       // Bn*Kn exactly
  atomicAdd(&deg[knn[e]], 1.0f);
}
__global__ __launch_bounds__(256) void dinv_kernel(
    const float* __restrict__ deg, float* __restrict__ dinv) {
  int i = blockIdx.x * 256 + threadIdx.x;
  dinv[i] = rsqrtf(deg[i]);
}

// ---------------------------------------------------------------------------
// 6) f32 -> bf16 weight conversion
// ---------------------------------------------------------------------------
__global__ __launch_bounds__(256) void cvt_bf16_kernel(
    const float* __restrict__ in, __bf16* __restrict__ out, int n) {
  int i = blockIdx.x * 256 + threadIdx.x;
  if (i < n) out[i] = f2bf(in[i]);
}

// ---------------------------------------------------------------------------
// 7) bf16 WMMA GEMM:  C(f32) = A(bf16, lda) x B(bf16, ldb row-major KxN)
//    Block tile 128x64, BK=32, 8 waves (4x2), wave tile 32x32.
// ---------------------------------------------------------------------------
template <bool BIAS, bool RELU>
__global__ __launch_bounds__(256) void gemm_bf16_kernel(
    const __bf16* __restrict__ A, int lda, const __bf16* __restrict__ Bw,
    int ldb, float* __restrict__ C, int ldc, const float* __restrict__ bias,
    int K) {
  __shared__ __bf16 As[128][40];   // [m][k], pad 32->40
  __shared__ __bf16 Bs[64][40];    // transposed: [n][k]

  int tid  = threadIdx.x;
  int lane = tid & 31;
  int wave = tid >> 5;
  int wm = (wave & 3) * 32;
  int wn = (wave >> 2) * 32;
  int bm0 = blockIdx.y * 128;
  int bn0 = blockIdx.x * 64;

  v8f acc[2][2];
#pragma unroll
  for (int i = 0; i < 2; ++i)
#pragma unroll
    for (int j = 0; j < 2; ++j)
#pragma unroll
      for (int e = 0; e < 8; ++e) acc[i][j][e] = 0.f;

  int rl = lane & 15, kh = lane >> 4;

  for (int k0 = 0; k0 < K; k0 += 32) {
#pragma unroll
    for (int i = 0; i < 8; ++i) {
      int idx = tid + i * 256;
      int r = idx >> 4, cu = idx & 15;
      unsigned u = *(const unsigned*)(A + (size_t)(bm0 + r) * lda + k0 + 2 * cu);
      *(unsigned*)(&As[r][2 * cu]) = u;
    }
#pragma unroll
    for (int i = 0; i < 8; ++i) {
      int idx = tid + i * 256;
      int kr = idx >> 6, n = idx & 63;
      Bs[n][kr] = Bw[(size_t)(k0 + kr) * ldb + bn0 + n];
    }
    __syncthreads();

    v16bf af[2], bfv[2];
    af[0]  = ldfrag_bf16(&As[wm + rl][0], kh);
    af[1]  = ldfrag_bf16(&As[wm + 16 + rl][0], kh);
    bfv[0] = ldfrag_bf16(&Bs[wn + rl][0], kh);
    bfv[1] = ldfrag_bf16(&Bs[wn + 16 + rl][0], kh);
#pragma unroll
    for (int i = 0; i < 2; ++i)
#pragma unroll
      for (int j = 0; j < 2; ++j)
        acc[i][j] = __builtin_amdgcn_wmma_f32_16x16x32_bf16(
            false, af[i], false, bfv[j], (short)0, acc[i][j], false, false);
    __syncthreads();
  }

#pragma unroll
  for (int i = 0; i < 2; ++i)
#pragma unroll
    for (int j = 0; j < 2; ++j)
#pragma unroll
      for (int r = 0; r < 8; ++r) {
        int row = bm0 + wm + i * 16 + r + 8 * kh;
        int col = bn0 + wn + j * 16 + rl;
        float v = acc[i][j][r];
        if (BIAS) v += bias[col];
        if (RELU) v = v > 0.f ? v : 0.f;
        C[(size_t)row * ldc + col] = v;
      }
}

// ---------------------------------------------------------------------------
// 8) GCN aggregation: self term (also initializes out), then edge scatter
// ---------------------------------------------------------------------------
__global__ __launch_bounds__(256) void agg_self_kernel(
    const float* __restrict__ z, const float* __restrict__ dinv,
    float* __restrict__ out) {
  int idx = blockIdx.x * 256 + threadIdx.x;   // over Bn*Hd
  int i = idx >> 8;                           // Hd == 256
  out[idx] = dinv[i] * dinv[i] * z[idx];
}
__global__ __launch_bounds__(256) void agg_edge_kernel(
    const float* __restrict__ z, const int* __restrict__ knn,
    const float* __restrict__ dinv, float* __restrict__ out) {
  int e = blockIdx.x;          // Bn*Kn edges
  int i = e / Kn;
  int j = knn[e];
  float w = dinv[i] * dinv[j];
  int f = threadIdx.x;         // Hd threads
  atomicAdd(&out[(size_t)j * Hd + f], w * z[(size_t)i * Hd + f]);
}

// ---------------------------------------------------------------------------
// 9) post-aggregation bias + BN (+ReLU), emits bf16 for the next GEMM
// ---------------------------------------------------------------------------
__global__ __launch_bounds__(256) void bnpost_kernel(
    const float* __restrict__ agg, const float* __restrict__ bias,
    const float* __restrict__ g, const float* __restrict__ b,
    const float* __restrict__ m, const float* __restrict__ v,
    __bf16* __restrict__ out, int ostride, int relu) {
  int idx = blockIdx.x * 256 + threadIdx.x;   // over Bn*Hd
  int c = idx & (Hd - 1);
  int r = idx >> 8;
  float val = agg[idx] + bias[c];
  val = (val - m[c]) * rsqrtf(v[c] + EPSF) * g[c] + b[c];
  if (relu) val = val > 0.f ? val : 0.f;
  out[(size_t)r * ostride + c] = f2bf(val);
}

// ---------------------------------------------------------------------------
// 10) final small GEMM: out = hid @ Wc2 + bc2  (128 x 38, fp32 VALU)
// ---------------------------------------------------------------------------
__global__ __launch_bounds__(64) void final_kernel(
    const float* __restrict__ hid, const float* __restrict__ Wc2,
    const float* __restrict__ bc2, float* __restrict__ out) {
  __shared__ float h[HC];
  int r = blockIdx.x;
  int t = threadIdx.x;
  h[t]      = hid[(size_t)r * HC + t];
  h[t + 64] = hid[(size_t)r * HC + t + 64];
  __syncthreads();
  if (t < Cd) {
    float s = bc2[t];
#pragma unroll 8
    for (int k = 0; k < HC; ++k) s += h[k] * Wc2[k * Cd + t];
    out[(size_t)r * Cd + t] = s;
  }
}

// ---------------------------------------------------------------------------
// Host-side orchestration
// ---------------------------------------------------------------------------
static inline size_t alignup(size_t x) { return (x + 255) & ~(size_t)255; }

extern "C" void kernel_launch(void* const* d_in, const int* in_sizes, int n_in,
                              void* d_out, int out_size, void* d_ws,
                              size_t ws_size, hipStream_t stream) {
  (void)in_sizes; (void)n_in; (void)out_size; (void)ws_size;

  const float* feat  = (const float*)d_in[0];
  const float* noise = (const float*)d_in[1];
  const float* bnf_g = (const float*)d_in[2];
  const float* bnf_b = (const float*)d_in[3];
  const float* bnf_m = (const float*)d_in[4];
  const float* bnf_v = (const float*)d_in[5];
  const float* W1 = (const float*)d_in[6];
  const float* b1 = (const float*)d_in[7];
  const float* W2 = (const float*)d_in[8];
  const float* b2 = (const float*)d_in[9];
  const float* W3 = (const float*)d_in[10];
  const float* b3 = (const float*)d_in[11];
  const float* bn1_g = (const float*)d_in[12];
  const float* bn1_b = (const float*)d_in[13];
  const float* bn1_m = (const float*)d_in[14];
  const float* bn1_v = (const float*)d_in[15];
  const float* bn2_g = (const float*)d_in[16];
  const float* bn2_b = (const float*)d_in[17];
  const float* bn2_m = (const float*)d_in[18];
  const float* bn2_v = (const float*)d_in[19];
  const float* bn3_g = (const float*)d_in[20];
  const float* bn3_b = (const float*)d_in[21];
  const float* bn3_m = (const float*)d_in[22];
  const float* bn3_v = (const float*)d_in[23];
  const float* Wc1 = (const float*)d_in[24];
  const float* bc1 = (const float*)d_in[25];
  const float* Wc2 = (const float*)d_in[26];
  const float* bc2 = (const float*)d_in[27];
  float* out = (float*)d_out;

  // workspace carve-up
  char* p = (char*)d_ws;
  size_t off = 0;
  auto take = [&](size_t bytes) {
    char* q = p + off;
    off = alignup(off + bytes);
    return q;
  };
  float*  dmat  = (float*)take((size_t)Bn * Bn * 4);
  __bf16* comb  = (__bf16*)take((size_t)Bn * CB * 2);   // [h3 | x_hi]
  __bf16* xlo   = (__bf16*)take((size_t)Bn * Fd * 2);   // bf16 split residual
  __bf16* w1bf  = (__bf16*)take((size_t)Fd * Hd * 2);
  __bf16* w2bf  = (__bf16*)take((size_t)Hd * Hd * 2);
  __bf16* w3bf  = (__bf16*)take((size_t)Hd * Hd * 2);
  __bf16* wc1bf = (__bf16*)take((size_t)CB * HC * 2);
  float*  z     = (float*)take((size_t)Bn * Hd * 4);
  float*  agg   = (float*)take((size_t)Bn * Hd * 4);
  __bf16* h1bf  = (__bf16*)take((size_t)Bn * Hd * 2);
  __bf16* h2bf  = (__bf16*)take((size_t)Bn * Hd * 2);
  float*  hid   = (float*)take((size_t)Bn * HC * 4);
  float*  sq    = (float*)take((size_t)Bn * 4);
  float*  deg   = (float*)take((size_t)Bn * 4);
  float*  dinv  = (float*)take((size_t)Bn * 4);
  int*    knn   = (int*)take((size_t)Bn * Kn * 4);

  // 1) feature BN -> comb[:,256:] (hi) + xlo (lo)
  bnf_kernel<<<(Bn * Fd) / 256, 256, 0, stream>>>(feat, bnf_g, bnf_b, bnf_m,
                                                  bnf_v, comb, xlo);
  // 2) row norms (fp32, recomputed from features)
  rowsq_kernel<<<Bn, 256, 0, stream>>>(feat, bnf_g, bnf_b, bnf_m, bnf_v, sq);
  // 3) split-bf16 WMMA Gram + distances
  gram_dist_kernel<<<dim3(Bn / 64, Bn / 128), 256, 0, stream>>>(
      comb + Hd, CB, xlo, sq, noise, dmat);
  // 4) top-k neighbors (single pass over dmat)
  topk_kernel<<<Bn, 256, 0, stream>>>(dmat, knn);
  // 5) degrees
  deg_init_kernel<<<Bn / 256, 256, 0, stream>>>(deg);
  deg_edge_kernel<<<(Bn * Kn) / 256, 256, 0, stream>>>(knn, deg);
  dinv_kernel<<<Bn / 256, 256, 0, stream>>>(deg, dinv);
  // 6) weight conversions
  cvt_bf16_kernel<<<(Fd * Hd + 255) / 256, 256, 0, stream>>>(W1, w1bf, Fd * Hd);
  cvt_bf16_kernel<<<(Hd * Hd + 255) / 256, 256, 0, stream>>>(W2, w2bf, Hd * Hd);
  cvt_bf16_kernel<<<(Hd * Hd + 255) / 256, 256, 0, stream>>>(W3, w3bf, Hd * Hd);
  cvt_bf16_kernel<<<(CB * HC + 255) / 256, 256, 0, stream>>>(Wc1, wc1bf,
                                                             CB * HC);

  // ---- GCN layer 1: z = x @ W1 (A = comb[:,256:], lda=CB)
  gemm_bf16_kernel<false, false><<<dim3(Hd / 64, Bn / 128), 256, 0, stream>>>(
      comb + Hd, CB, w1bf, Hd, z, Hd, nullptr, Fd);
  agg_self_kernel<<<(Bn * Hd) / 256, 256, 0, stream>>>(z, dinv, agg);
  agg_edge_kernel<<<Bn * Kn, Hd, 0, stream>>>(z, knn, dinv, agg);
  bnpost_kernel<<<(Bn * Hd) / 256, 256, 0, stream>>>(
      agg, b1, bn1_g, bn1_b, bn1_m, bn1_v, h1bf, Hd, 1);

  // ---- GCN layer 2
  gemm_bf16_kernel<false, false><<<dim3(Hd / 64, Bn / 128), 256, 0, stream>>>(
      h1bf, Hd, w2bf, Hd, z, Hd, nullptr, Hd);
  agg_self_kernel<<<(Bn * Hd) / 256, 256, 0, stream>>>(z, dinv, agg);
  agg_edge_kernel<<<Bn * Kn, Hd, 0, stream>>>(z, knn, dinv, agg);
  bnpost_kernel<<<(Bn * Hd) / 256, 256, 0, stream>>>(
      agg, b2, bn2_g, bn2_b, bn2_m, bn2_v, h2bf, Hd, 1);

  // ---- GCN layer 3 (no relu), writes into comb[:, :256]
  gemm_bf16_kernel<false, false><<<dim3(Hd / 64, Bn / 128), 256, 0, stream>>>(
      h2bf, Hd, w3bf, Hd, z, Hd, nullptr, Hd);
  agg_self_kernel<<<(Bn * Hd) / 256, 256, 0, stream>>>(z, dinv, agg);
  agg_edge_kernel<<<Bn * Kn, Hd, 0, stream>>>(z, knn, dinv, agg);
  bnpost_kernel<<<(Bn * Hd) / 256, 256, 0, stream>>>(
      agg, b3, bn3_g, bn3_b, bn3_m, bn3_v, comb, CB, 0);

  // ---- classifier: hid = relu(comb @ Wc1 + bc1)
  gemm_bf16_kernel<true, true><<<dim3(HC / 64, Bn / 128), 256, 0, stream>>>(
      comb, CB, wc1bf, HC, hid, HC, bc1, CB);
  // ---- out = hid @ Wc2 + bc2
  final_kernel<<<Bn, 64, 0, stream>>>(hid, Wc2, bc2, out);
}
